// MultiRelGraphConv_42898133352617
// MI455X (gfx1250) — compile-verified, matrix operand order checked by reading, not run on hardware
//
#include <hip/hip_runtime.h>
#include <hip/hip_bf16.h>

// ---------------------------------------------------------------------------
// GNN layer pipeline on gfx1250: v_wmma_f32_16x16x32_bf16 GEMMs +
// global_load_async_to_lds_b128 double-buffered gather staging.
//   D = H = 128 fixed by the reference.
// ---------------------------------------------------------------------------

typedef __attribute__((ext_vector_type(16))) __bf16 v16bf;
typedef __attribute__((ext_vector_type(8)))  __bf16 v8bf;
typedef __attribute__((ext_vector_type(4)))  __bf16 v4bf;
typedef __attribute__((ext_vector_type(8)))  float  v8f;

#define RRELU_SLOPE 0.22916666666666666f

__device__ __forceinline__ v16bf ldfrag(const __bf16* p) {
    // per-lane fragment: elements 0..7 = K[klo..klo+7], 8..15 = K[klo+16..klo+23]
    v8bf lo = *(const v8bf*)p;
    v8bf hi = *(const v8bf*)(p + 16);
    return __builtin_shufflevector(lo, hi, 0,1,2,3,4,5,6,7,8,9,10,11,12,13,14,15);
}

__device__ __forceinline__ v8f wmma_bf16(v16bf a, v16bf b, v8f c) {
    return __builtin_amdgcn_wmma_f32_16x16x32_bf16(
        /*neg_a=*/false, a, /*neg_b=*/false, b,
        /*c_mod=*/(short)0, c, /*reuse_a=*/false, /*reuse_b=*/false);
}

// 16-byte async global -> LDS copy (per-lane addresses, tracked by ASYNCcnt)
__device__ __forceinline__ void async_copy16(unsigned lds_addr, const void* gptr) {
    asm volatile("global_load_async_to_lds_b128 %0, %1, off"
                 :: "v"(lds_addr), "v"(gptr) : "memory");
}

// ---------------------------------------------------------------------------
// utility kernels
// ---------------------------------------------------------------------------
__global__ __launch_bounds__(256) void zero_f32(float* p, long n) {
    long i = (long)blockIdx.x * blockDim.x + threadIdx.x;
    long stride = (long)gridDim.x * blockDim.x;
    for (; i < n; i += stride) p[i] = 0.0f;
}

__global__ __launch_bounds__(256) void count_edges(const int* __restrict__ dst,
                                                   float* __restrict__ cnt, int E) {
    int e = blockIdx.x * blockDim.x + threadIdx.x;
    if (e < E) atomicAdd(&cnt[dst[e]], 1.0f);
}

// vectorized fp32 -> bf16 conversion (n4 = element count / 4)
__global__ __launch_bounds__(256) void cvt_f32_bf16(const float* __restrict__ in,
                                                    __bf16* __restrict__ out, long n4) {
    long i = (long)blockIdx.x * blockDim.x + threadIdx.x;
    long stride = (long)gridDim.x * blockDim.x;
    for (; i < n4; i += stride) {
        float4 v = ((const float4*)in)[i];
        v4bf o = { (__bf16)v.x, (__bf16)v.y, (__bf16)v.z, (__bf16)v.w };
        ((v4bf*)out)[i] = o;
    }
}

// ---------------------------------------------------------------------------
// Edge GEMM: msg[e,:] = concat(hb[src[e]], ef[e]) @ W  (K=256, N=128)
// scatter-add into s[dst[e], :] with f32 atomics (bias folded into node pass).
// Block: 256 threads (8 waves). Tile: 32 edges x 128 outputs, double-buffered.
// ---------------------------------------------------------------------------
#define KE   256
#define ESTR (KE + 8)   // 264 bf16 => 528B row stride, conflict-free b128 frags

__global__ __launch_bounds__(256) void edge_gemm(
    const __bf16* __restrict__ hb, const float* __restrict__ ef,
    const int* __restrict__ src, const int* __restrict__ dstv,
    const float* __restrict__ W, float* __restrict__ sbuf,
    long E, int numTiles)
{
    __shared__ __bf16 Wt[128 * ESTR];     // Wt[n][k]  (B supplied as W^T rows)
    __shared__ __bf16 Abuf[2][32 * ESTR]; // double-buffered A tiles

    const int t = threadIdx.x;

    // Stage W^T into LDS once per block (converted to bf16).
    for (int idx = t; idx < 256 * 128; idx += 256) {
        int k = idx >> 7, n = idx & 127;
        Wt[n * ESTR + k] = (__bf16)W[idx];
    }

    const int row = t >> 3;           // 8 staging threads per edge row
    const int sub = t & 7;
    const unsigned ldsA[2] = { (unsigned)(uintptr_t)&Abuf[0][0],
                               (unsigned)(uintptr_t)&Abuf[1][0] };

    // Stage one 32-edge tile into Abuf[buf]:
    //  cols   0..127: bf16 gather rows via async global->LDS b128 (2 issues/lane)
    //  cols 128..255: streaming fp32 edge feats, cvt->bf16, plain LDS stores
    auto stage = [&](long tile, int buf) {
        long e  = tile * 32 + row;
        long ec = e < E ? e : E - 1;          // clamp tail: garbage row is discarded
        int  sidx = src[ec];
        const __bf16* g0 = hb + (size_t)sidx * 128 + sub * 16;
        unsigned l0 = ldsA[buf] + (unsigned)(row * ESTR + sub * 16) * 2u;
        async_copy16(l0,       g0);
        async_copy16(l0 + 16u, g0 + 8);
        const float4* p4 = (const float4*)(ef + (size_t)ec * 128 + sub * 16);
        __bf16* d = &Abuf[buf][row * ESTR + 128 + sub * 16];
        #pragma unroll
        for (int j = 0; j < 4; ++j) {
            float4 v = p4[j];
            d[j * 4 + 0] = (__bf16)v.x;
            d[j * 4 + 1] = (__bf16)v.y;
            d[j * 4 + 2] = (__bf16)v.z;
            d[j * 4 + 3] = (__bf16)v.w;
        }
    };

    const int lane = t & 31;
    const int wave = t >> 5;
    const int mt   = wave >> 2;         // 0..1  (16-row block of the 32-edge tile)
    const int nt0  = (wave & 3) * 2;    // this wave owns N-tiles nt0, nt0+1
    const int klo  = (lane & 16) ? 8 : 0;
    const int l15  = lane & 15;

    long tile = blockIdx.x;
    if (tile < numTiles) stage(tile, 0);
    int cur = 0;

    for (; tile < numTiles; tile += gridDim.x) {
        long nxt = tile + gridDim.x;
        if (nxt < numTiles) {
            stage(nxt, cur ^ 1);                       // prefetch next tile
            asm volatile("s_wait_asynccnt 0x2" ::: "memory"); // cur tile landed
        } else {
            asm volatile("s_wait_asynccnt 0x0" ::: "memory");
        }
        __syncthreads();   // cur buffer (async h-part + sync ef-part + Wt) visible

        // ---- WMMA K-loop on Abuf[cur] ----
        v8f acc0 = {}; v8f acc1 = {};
        const __bf16* ap  = &Abuf[cur][(mt * 16 + l15) * ESTR + klo];
        const __bf16* bp0 = &Wt[(nt0 * 16 + l15) * ESTR + klo];
        const __bf16* bp1 = &Wt[((nt0 + 1) * 16 + l15) * ESTR + klo];
        #pragma unroll
        for (int kk = 0; kk < KE; kk += 32) {
            v16bf a  = ldfrag(ap  + kk);
            v16bf b0 = ldfrag(bp0 + kk);
            v16bf b1 = ldfrag(bp1 + kk);
            acc0 = wmma_bf16(a, b0, acc0);
            acc1 = wmma_bf16(a, b1, acc1);
        }

        // ---- epilogue: scatter-add into s[dst] ----
        const long mbase = tile * 32 + mt * 16 + ((lane & 16) ? 8 : 0);
        const int  col0  = nt0 * 16 + l15;
        #pragma unroll
        for (int g = 0; g < 8; ++g) {
            long e2 = mbase + g;
            if (e2 < E) {
                int d = dstv[e2];
                atomicAdd(&sbuf[(size_t)d * 128 + col0],      acc0[g]);
                atomicAdd(&sbuf[(size_t)d * 128 + col0 + 16], acc1[g]);
            }
        }
        __syncthreads();   // all waves done reading Abuf[cur]; safe to restage it
        cur ^= 1;
    }
}

// ---------------------------------------------------------------------------
// Node update: h_out = rrelu( where(cnt>0, s/cnt + b, 0) + h_in@L + lb )
// K=128, N=128. Tile: 32 nodes.
// ---------------------------------------------------------------------------
#define KN   128
#define NSTR (KN + 8)   // 136 bf16 => 272B stride

__global__ __launch_bounds__(256) void node_update(
    const float* __restrict__ hin, const float* __restrict__ L,
    const float* __restrict__ sbuf, const float* __restrict__ cnt,
    const float* __restrict__ bvec, const float* __restrict__ lbvec,
    float* __restrict__ hout, int N, int numTiles)
{
    __shared__ __bf16 Lt[128 * NSTR];
    __shared__ __bf16 At[32 * NSTR];

    const int t = threadIdx.x;
    for (int idx = t; idx < 128 * 128; idx += 256) {
        int k = idx >> 7, n = idx & 127;
        Lt[n * NSTR + k] = (__bf16)L[idx];
    }
    __syncthreads();

    const int lane = t & 31;
    const int wave = t >> 5;
    const int mt   = wave >> 2;
    const int nt0  = (wave & 3) * 2;
    const int klo  = (lane & 16) ? 8 : 0;
    const int l15  = lane & 15;

    const int arow = t >> 3;
    const int cchunk = (t & 7) * 16;    // 16 cols per thread

    for (int tile = blockIdx.x; tile < numTiles; tile += gridDim.x) {
        long node = (long)tile * 32 + arow;
        __bf16* arow_p = &At[arow * NSTR + cchunk];
        if (node < N) {
            const float4* p4 = (const float4*)(hin + (size_t)node * 128 + cchunk);
            #pragma unroll
            for (int j = 0; j < 4; ++j) {
                float4 v = p4[j];
                arow_p[j * 4 + 0] = (__bf16)v.x;
                arow_p[j * 4 + 1] = (__bf16)v.y;
                arow_p[j * 4 + 2] = (__bf16)v.z;
                arow_p[j * 4 + 3] = (__bf16)v.w;
            }
        } else {
            #pragma unroll
            for (int j = 0; j < 16; ++j) arow_p[j] = (__bf16)0.0f;
        }
        __syncthreads();

        v8f acc0 = {}; v8f acc1 = {};
        const __bf16* ap  = &At[(mt * 16 + l15) * NSTR + klo];
        const __bf16* bp0 = &Lt[(nt0 * 16 + l15) * NSTR + klo];
        const __bf16* bp1 = &Lt[((nt0 + 1) * 16 + l15) * NSTR + klo];
        #pragma unroll
        for (int kk = 0; kk < KN; kk += 32) {
            v16bf a  = ldfrag(ap  + kk);
            v16bf b0 = ldfrag(bp0 + kk);
            v16bf b1 = ldfrag(bp1 + kk);
            acc0 = wmma_bf16(a, b0, acc0);
            acc1 = wmma_bf16(a, b1, acc1);
        }
        __syncthreads();

        const long mbase = (long)tile * 32 + mt * 16 + ((lane & 16) ? 8 : 0);
        const int  col0  = nt0 * 16 + l15;
        const float bb0 = bvec[col0],  bb1 = bvec[col0 + 16];
        const float lb0 = lbvec[col0], lb1 = lbvec[col0 + 16];
        #pragma unroll
        for (int g = 0; g < 8; ++g) {
            long nd = mbase + g;
            if (nd < N) {
                float c = cnt[nd];
                float r0 = acc0[g] + lb0;
                float r1 = acc1[g] + lb1;
                if (c > 0.0f) {
                    float inv = 1.0f / c;
                    r0 += sbuf[(size_t)nd * 128 + col0]      * inv + bb0;
                    r1 += sbuf[(size_t)nd * 128 + col0 + 16] * inv + bb1;
                }
                r0 = (r0 >= 0.0f) ? r0 : r0 * RRELU_SLOPE;
                r1 = (r1 >= 0.0f) ? r1 : r1 * RRELU_SLOPE;
                hout[(size_t)nd * 128 + col0]      = r0;
                hout[(size_t)nd * 128 + col0 + 16] = r1;
            }
        }
    }
}

// ---------------------------------------------------------------------------
// Output GEMM: out = concat(h1,h2) @ Wo + bo   (K=256, N=128)
// ---------------------------------------------------------------------------
__global__ __launch_bounds__(256) void out_gemm(
    const float* __restrict__ h1, const float* __restrict__ h2,
    const float* __restrict__ Wo, const float* __restrict__ bo,
    float* __restrict__ out, int N, int numTiles)
{
    __shared__ __bf16 Wt[128 * ESTR];
    __shared__ __bf16 At[32 * ESTR];

    const int t = threadIdx.x;
    for (int idx = t; idx < 256 * 128; idx += 256) {
        int k = idx >> 7, n = idx & 127;
        Wt[n * ESTR + k] = (__bf16)Wo[idx];
    }
    __syncthreads();

    const int lane = t & 31;
    const int wave = t >> 5;
    const int mt   = wave >> 2;
    const int nt0  = (wave & 3) * 2;
    const int klo  = (lane & 16) ? 8 : 0;
    const int l15  = lane & 15;

    const int arow = t >> 3;
    const int cchunk = (t & 7) * 32;

    for (int tile = blockIdx.x; tile < numTiles; tile += gridDim.x) {
        long node = (long)tile * 32 + arow;
        __bf16* arow_p = &At[arow * ESTR + cchunk];
        if (node < N) {
            const float4* p4 = (cchunk < 128)
                ? (const float4*)(h1 + (size_t)node * 128 + cchunk)
                : (const float4*)(h2 + (size_t)node * 128 + (cchunk - 128));
            #pragma unroll
            for (int j = 0; j < 8; ++j) {
                float4 v = p4[j];
                arow_p[j * 4 + 0] = (__bf16)v.x;
                arow_p[j * 4 + 1] = (__bf16)v.y;
                arow_p[j * 4 + 2] = (__bf16)v.z;
                arow_p[j * 4 + 3] = (__bf16)v.w;
            }
        } else {
            #pragma unroll
            for (int j = 0; j < 32; ++j) arow_p[j] = (__bf16)0.0f;
        }
        __syncthreads();

        v8f acc0 = {}; v8f acc1 = {};
        const __bf16* ap  = &At[(mt * 16 + l15) * ESTR + klo];
        const __bf16* bp0 = &Wt[(nt0 * 16 + l15) * ESTR + klo];
        const __bf16* bp1 = &Wt[((nt0 + 1) * 16 + l15) * ESTR + klo];
        #pragma unroll
        for (int kk = 0; kk < KE; kk += 32) {
            v16bf a  = ldfrag(ap  + kk);
            v16bf b0 = ldfrag(bp0 + kk);
            v16bf b1 = ldfrag(bp1 + kk);
            acc0 = wmma_bf16(a, b0, acc0);
            acc1 = wmma_bf16(a, b1, acc1);
        }
        __syncthreads();

        const long mbase = (long)tile * 32 + mt * 16 + ((lane & 16) ? 8 : 0);
        const int  col0  = nt0 * 16 + l15;
        const float bo0 = bo[col0], bo1 = bo[col0 + 16];
        #pragma unroll
        for (int g = 0; g < 8; ++g) {
            long nd = mbase + g;
            if (nd < N) {
                out[(size_t)nd * 128 + col0]      = acc0[g] + bo0;
                out[(size_t)nd * 128 + col0 + 16] = acc1[g] + bo1;
            }
        }
    }
}

// ---------------------------------------------------------------------------
// host launcher
// ---------------------------------------------------------------------------
extern "C" void kernel_launch(void* const* d_in, const int* in_sizes, int n_in,
                              void* d_out, int out_size, void* d_ws, size_t ws_size,
                              hipStream_t stream) {
    (void)n_in; (void)out_size; (void)ws_size;

    const float* node_feats = (const float*)d_in[0];
    const float* edge_feats = (const float*)d_in[1];
    const int*   src        = (const int*)  d_in[2];
    const int*   dst        = (const int*)  d_in[3];
    const float* W1  = (const float*)d_in[4];
    const float* b1  = (const float*)d_in[5];
    const float* L1  = (const float*)d_in[6];
    const float* lb1 = (const float*)d_in[7];
    const float* W2  = (const float*)d_in[8];
    const float* b2  = (const float*)d_in[9];
    const float* L2  = (const float*)d_in[10];
    const float* lb2 = (const float*)d_in[11];
    const float* Wo  = (const float*)d_in[12];
    const float* bo  = (const float*)d_in[13];
    float* out = (float*)d_out;

    const long N = in_sizes[0] / 128;
    const long E = in_sizes[2];

    // workspace: s[N*128] f32 | cnt[N] f32 | h1[N*128] f32 | h2[N*128] f32 | hb[N*128] bf16
    float* s   = (float*)d_ws;
    float* cnt = s + N * 128;
    float* h1  = cnt + N;
    float* h2  = h1 + N * 128;
    __bf16* hb = (__bf16*)(h2 + N * 128);

    const int numTilesE = (int)((E + 31) / 32);
    const int numTilesN = (int)((N + 31) / 32);
    const int edgeBlocks = numTilesE < 2048 ? numTilesE : 2048;
    const int nodeBlocks = numTilesN < 2048 ? numTilesN : 2048;

    // layer 1
    zero_f32<<<1024, 256, 0, stream>>>(s, N * 128 + N);  // zeroes s and cnt
    count_edges<<<(int)((E + 255) / 256), 256, 0, stream>>>(dst, cnt, (int)E);
    cvt_f32_bf16<<<1024, 256, 0, stream>>>(node_feats, hb, N * 128 / 4);
    edge_gemm<<<edgeBlocks, 256, 0, stream>>>(hb, edge_feats, src, dst, W1, s, E, numTilesE);
    node_update<<<nodeBlocks, 256, 0, stream>>>(node_feats, L1, s, cnt, b1, lb1, h1, (int)N, numTilesN);

    // layer 2 (reuse s and hb)
    zero_f32<<<1024, 256, 0, stream>>>(s, N * 128);
    cvt_f32_bf16<<<1024, 256, 0, stream>>>(h1, hb, N * 128 / 4);
    edge_gemm<<<edgeBlocks, 256, 0, stream>>>(hb, edge_feats, src, dst, W2, s, E, numTilesE);
    node_update<<<nodeBlocks, 256, 0, stream>>>(h1, L2, s, cnt, b2, lb2, h2, (int)N, numTilesN);

    // output projection
    out_gemm<<<nodeBlocks, 256, 0, stream>>>(h1, h2, Wo, bo, out, (int)N, numTilesN);
}
